// Edge_learnable_70531952935057
// MI455X (gfx1250) — matrix operand order repaired and checked by reading, not compiled, not run on hardware
//
#include <hip/hip_runtime.h>
#include <hip/hip_bf16.h>

typedef __attribute__((ext_vector_type(16))) _Float16 v16h;
typedef __attribute__((ext_vector_type(8)))  float    v8f;
typedef __attribute__((ext_vector_type(4)))  unsigned v4u;
typedef __attribute__((ext_vector_type(8)))  int      v8i;
typedef __attribute__((ext_vector_type(4)))  int      v4i;

#define NN        4096
#define HH        128
#define NTOT      (NN * NN)           // 16777216
#define MAXC2     262144              // bound on 2*drop_edge_num candidates

#if __has_builtin(__builtin_amdgcn_tensor_load_to_lds) && __has_builtin(__builtin_amdgcn_s_wait_tensorcnt)
#define USE_TDM 1
#else
#define USE_TDM 0
#endif

// ---------------- workspace layout (bytes) ----------------
#define OFF_BIG16    0u                         // 32MB: gT16 then adj16 (reused)
#define OFF_FEATT    33554432u                  // 1MB : featT f16 (128 x 4096)
#define OFF_DSTEMB   34603008u                  // 2MB : dst_emb f32 (4096 x 128)
#define OFF_CANDIDX  36700160u                  // 1MB : int  [MAXC2]
#define OFF_CANDVAL  37748736u                  // 1MB : f32  [MAXC2]
#define OFF_SEL      38797312u                  // 1MB : int  [MAXC2]
#define OFF_TOP8     39845888u                  // 128KB: int [4096*8]
#define OFF_PRESENT  39976960u                  // 16KB: int [4096]
#define OFF_LAST     39993344u                  // 16KB: int [4096]
#define OFF_CTRL     40009728u                  // 2 x Ctrl
#define WS_NEEDED    40013824u

struct Ctrl {
  unsigned hist[256];
  unsigned prefix;   // resolved key bits (MSB-first)
  unsigned rank;     // remaining rank within current bucket (1-indexed)
  unsigned target;   // number of elements to select
  unsigned n;        // element count (mode 1)
  unsigned cnt_lo, cnt_eq;
  float    maxv, sumexp;
};

// ---------------- helpers ----------------
__device__ inline unsigned fkey(float f) {         // monotone float -> uint (ascending)
  unsigned u = __float_as_uint(f);
  return (u & 0x80000000u) ? ~u : (u | 0x80000000u);
}
__device__ inline float gumbelf(unsigned s) {
  s ^= s >> 16; s *= 0x7feb352du; s ^= s >> 15; s *= 0x846ca68bu; s ^= s >> 16;
  float u = ((float)s + 0.5f) * (1.0f / 4294967296.0f);
  return -__logf(-__logf(u));
}
__device__ inline void atomicMaxFloat(float* addr, float val) {
  unsigned* a = (unsigned*)addr;
  unsigned old = *a;
  while (__uint_as_float(old) < val) {
    unsigned assumed = old;
    old = atomicCAS(a, assumed, __float_as_uint(val));
    if (old == assumed) break;
  }
}

// ---------------- WMMA fragment loaders (CDNA5 16x16x32 f16 layouts) ----------------
// A 16x32 f16: lane L -> row m0+(L&15); VGPR0-3 = K g*8..g*8+7, VGPR4-7 = K 16+g*8.. (g=L>>4)
__device__ inline v16h load_A_h(const _Float16* A, int lda, int m0, int k0, int lane) {
  const _Float16* p = A + (size_t)(m0 + (lane & 15)) * lda + k0;
  int g = lane >> 4;
  v16h a;
#pragma unroll
  for (int i = 0; i < 8; ++i) a[i] = p[g * 8 + i];
#pragma unroll
  for (int i = 0; i < 8; ++i) a[8 + i] = p[16 + g * 8 + i];
  return a;
}
__device__ inline v16h load_A_f32(const float* A, int lda, int m0, int k0, int lane) {
  const float* p = A + (size_t)(m0 + (lane & 15)) * lda + k0;
  int g = lane >> 4;
  v16h a;
#pragma unroll
  for (int i = 0; i < 8; ++i) a[i] = (_Float16)p[g * 8 + i];
#pragma unroll
  for (int i = 0; i < 8; ++i) a[8 + i] = (_Float16)p[16 + g * 8 + i];
  return a;
}
// B 32x16 f16: lane L -> col n0+(L&15); K = g*8+i (i<8), 16+g*8+(i-8) (g=L>>4).  B[k][n] = R[n][k] for row-major R.
__device__ inline v16h load_B_rows(const _Float16* R, int ldr, int k0, int n0, int lane) {
  const _Float16* p = R + (size_t)(n0 + (lane & 15)) * ldr + k0 + (lane >> 4) * 8;
  v16h b;
#pragma unroll
  for (int i = 0; i < 8; ++i) b[i] = p[i];
#pragma unroll
  for (int i = 0; i < 8; ++i) b[8 + i] = p[16 + i];
  return b;
}
__device__ inline v16h load_B_f32s(const float* W, int ldw, int k0, int n0, int lane) {
  int c = n0 + (lane & 15), g = lane >> 4;
  v16h b;
#pragma unroll
  for (int i = 0; i < 16; ++i) b[i] = (_Float16)W[(size_t)(k0 + g * 8 + (i & 7) + (i >> 3) * 16) * ldw + c];
  return b;
}
__device__ inline v8f wmma16(v16h a, v16h b, v8f c) {
  return __builtin_amdgcn_wmma_f32_16x16x32_f16(false, a, false, b, (short)0, c, false, false);
}

#if USE_TDM
// Issue a TDM 2-D tile load: tile_h rows x tile_w f16 elements, row stride
// stride_elems in global memory, packed contiguously (tile_w per row) into LDS.
// D# bit layout per CDNA5 ISA ch.8 (group0 128b, group1 256b, groups2/3 zero = 2D).
// clang-23 builtin arity: (v4u g0, v8i g1, v4i g2, v4i g3, v8i extra, i32 cpol).
__device__ inline void tdm_load_2d(unsigned lds_off, const _Float16* gptr,
                                   unsigned tile_w, unsigned tile_h,
                                   unsigned stride_elems) {
  unsigned long long ga = (unsigned long long)(size_t)gptr;
  v4u g0;
  g0[0] = 1u;                                            // count=1 (user mode)
  g0[1] = lds_off;                                       // lds_addr (bytes)
  g0[2] = (unsigned)(ga & 0xFFFFFFFFu);                  // global_addr[31:0]
  g0[3] = (unsigned)((ga >> 32) & 0x01FFFFFFu)           // global_addr[56:32]
          | (2u << 30);                                  // type=2 ("image")
  v8i g1;
  g1[0] = (int)(1u << 16);                               // wg_mask=0, data_size=1 (2B)
  g1[1] = (int)((tile_w & 0xFFFFu) << 16);               // tensor_dim0[15:0] (=tile_w)
  g1[2] = (int)(((tile_w >> 16) & 0xFFFFu) |             // tensor_dim0[31:16]
                ((tile_h & 0xFFFFu) << 16));             // tensor_dim1[15:0] (=tile_h)
  g1[3] = (int)(((tile_h >> 16) & 0xFFFFu) |             // tensor_dim1[31:16]
                ((tile_w & 0xFFFFu) << 16));             // tile_dim0
  g1[4] = (int)(tile_h & 0xFFFFu);                       // tile_dim1, tile_dim2=0
  g1[5] = (int)stride_elems;                             // tensor_dim0_stride[31:0]
  g1[6] = 0;                                             // stride[47:32], dim1_stride lo
  g1[7] = 0;
  v4i z4 = {0, 0, 0, 0};
  v8i z8 = {0, 0, 0, 0, 0, 0, 0, 0};
  __builtin_amdgcn_tensor_load_to_lds(g0, g1, z4, z4, z8, 0);
}
#endif

// ==================== kernels ====================

// feat = relu(src @ ft_w + ft_b), stored TRANSPOSED as f16: featT[n*4096 + m]
__global__ void k_feat(const float* __restrict__ src, const float* __restrict__ w,
                       const float* __restrict__ b, _Float16* __restrict__ featT) {
  int lane = threadIdx.x & 31, wave = threadIdx.x >> 5;
  int tid = blockIdx.x * 8 + wave;          // 2048 tiles = 256 (M) x 8 (N)
  int m0 = (tid >> 3) * 16, n0 = (tid & 7) * 16;
  v8f c = {};
  for (int k0 = 0; k0 < HH; k0 += 32) {
    v16h a = load_A_f32(src, HH, m0, k0, lane);
    v16h bb = load_B_f32s(w, HH, k0, n0, lane);
    c = wmma16(a, bb, c);
  }
  int col = n0 + (lane & 15);
  int rb = m0 + (lane >> 4) * 8;
  float bias = b[col];
#pragma unroll
  for (int v = 0; v < 8; ++v) {
    float x = fmaxf(c[v] + bias, 0.f);
    featT[(size_t)col * NN + rb + v] = (_Float16)x;
  }
}

// gT16[m*4096+k] = (f16) sub_g[k*4096+m]   (LDS-tiled transpose+convert)
__global__ void k_transpose(const float* __restrict__ g, _Float16* __restrict__ gT) {
  __shared__ float t[32][33];
  int tx = threadIdx.x, ty = threadIdx.y;          // (32, 8)
  int bx = blockIdx.x * 32, by = blockIdx.y * 32;
#pragma unroll
  for (int j = 0; j < 4; ++j)
    t[ty + j * 8][tx] = g[(size_t)(by + ty + j * 8) * NN + bx + tx];
  __syncthreads();
#pragma unroll
  for (int j = 0; j < 4; ++j)
    gT[(size_t)(bx + ty + j * 8) * NN + by + tx] = (_Float16)t[tx][ty + j * 8];
}

// dst_emb = gT16 @ feat   (A = gT16 rows, B = featT rows), output f32 (4096 x 128)
__global__ void k_dstemb(const _Float16* __restrict__ gT, const _Float16* __restrict__ featT,
                         float* __restrict__ dst) {
  int lane = threadIdx.x & 31, wave = threadIdx.x >> 5;
  int tid = blockIdx.x * 8 + wave;          // 2048 tiles = 256 x 8
  int m0 = (tid >> 3) * 16, n0 = (tid & 7) * 16;
  v8f c = {};
  for (int k0 = 0; k0 < NN; k0 += 32) {
    v16h a = load_A_h(gT, NN, m0, k0, lane);
    v16h b = load_B_rows(featT, NN, k0, n0, lane);
    c = wmma16(a, b, c);
  }
  int col = n0 + (lane & 15);
  int rb = m0 + (lane >> 4) * 8;
#pragma unroll
  for (int v = 0; v < 8; ++v) dst[(size_t)(rb + v) * HH + col] = c[v];
}

// ---- radix-select machinery ----
__global__ void k_init_ctrl(Ctrl* c0, Ctrl* c1, const int* __restrict__ pK) {
  int t = threadIdx.x;
  if (t < 256) { c0->hist[t] = 0; c1->hist[t] = 0; }
  if (t == 0) {
    unsigned K = (unsigned)max(1, *pK);
    unsigned E2 = min(2u * K, (unsigned)MAXC2);
    if (K > E2) K = E2;
    c0->prefix = 0; c0->rank = E2; c0->target = E2; c0->n = NTOT;
    c0->cnt_lo = 0; c0->cnt_eq = 0; c0->maxv = -3.0e38f; c0->sumexp = 0.f;
    c1->prefix = 0; c1->rank = K;  c1->target = K;  c1->n = E2;
    c1->cnt_lo = 0; c1->cnt_eq = 0; c1->maxv = -3.0e38f; c1->sumexp = 0.f;
  }
}

__global__ void k_init_arrays(int* sel, int* present, int* last) {
  for (int i = blockIdx.x * blockDim.x + threadIdx.x; i < MAXC2; i += gridDim.x * blockDim.x) {
    sel[i] = 0;
    if (i < NN) { present[i] = 0; last[i] = -1; }
  }
}

// mode 0: key = fkey(grad[i]*g[i]) over NTOT  (ascending = most negative first)
// mode 1: key = ~fkey(vals[i])     over c->n  (descending)
__global__ void k_hist(const float* __restrict__ grad, const float* __restrict__ g,
                       const float* __restrict__ vals, Ctrl* c, int mode,
                       unsigned maskHi, int shift, int n0) {
  __shared__ unsigned lh[256];
  if (threadIdx.x < 256) lh[threadIdx.x] = 0;
  __syncthreads();
  int n = mode ? (int)c->n : n0;
  unsigned pre = c->prefix & maskHi;
  for (long i = (long)blockIdx.x * blockDim.x + threadIdx.x; i < n;
       i += (long)gridDim.x * blockDim.x) {
    unsigned key = mode ? ~fkey(vals[i]) : fkey(grad[i] * g[i]);
    if ((key & maskHi) == pre) atomicAdd(&lh[(key >> shift) & 255u], 1u);
  }
  __syncthreads();
  if (threadIdx.x < 256 && lh[threadIdx.x]) atomicAdd(&c->hist[threadIdx.x], lh[threadIdx.x]);
}

__global__ void k_scan(Ctrl* c, int shift) {
  if (threadIdx.x == 0) {
    unsigned cum = 0, r = c->rank;
    for (int b = 0; b < 256; ++b) {
      unsigned h = c->hist[b];
      if (cum + h >= r) { c->prefix |= ((unsigned)b) << shift; c->rank = r - cum; break; }
      cum += h;
    }
  }
  __syncthreads();
  if (threadIdx.x < 256) c->hist[threadIdx.x] = 0;
}

__global__ void k_gather(const float* __restrict__ grad, const float* __restrict__ g,
                         const float* __restrict__ vals, Ctrl* c, int mode, int n0,
                         int* __restrict__ out_idx, int* __restrict__ out_sel) {
  int n = mode ? (int)c->n : n0;
  unsigned T = c->prefix, r = c->rank, tgt = c->target;
  for (long i = (long)blockIdx.x * blockDim.x + threadIdx.x; i < n;
       i += (long)gridDim.x * blockDim.x) {
    unsigned key = mode ? ~fkey(vals[i]) : fkey(grad[i] * g[i]);
    if (key < T) {
      if (mode) out_sel[i] = 1;
      else { unsigned s = atomicAdd(&c->cnt_lo, 1u); if (s < tgt) out_idx[s] = (int)i; }
    } else if (key == T) {
      unsigned t = atomicAdd(&c->cnt_eq, 1u);
      if (t < r) { if (mode) out_sel[i] = 1; else out_idx[tgt - r + t] = (int)i; }
    }
  }
}

// drop-branch MLP: logit(e) + gumbel -> cand_val[e]
__global__ void k_drop_mlp(const float* __restrict__ src_emb, const float* __restrict__ dst_emb,
                           const float* __restrict__ w1, const float* __restrict__ b1,
                           const float* __restrict__ w2, const float* __restrict__ b2,
                           const int* __restrict__ cand, float* __restrict__ cand_val,
                           const Ctrl* c0) {
  int e = blockIdx.x;
  if (e >= (int)c0->target) return;
  int idx = cand[e], s = idx >> 12, d = idx & (NN - 1);
  __shared__ float emb[2 * HH];
  __shared__ float red[HH];
  int t = threadIdx.x;                       // 128
  emb[t] = src_emb[(size_t)s * HH + t];
  emb[HH + t] = dst_emb[(size_t)d * HH + t];
  __syncthreads();
  float h = b1[t];
  for (int k = 0; k < 2 * HH; ++k) h = fmaf(emb[k], w1[(size_t)k * HH + t], h);
  h = fmaxf(h, 0.f);
  red[t] = h * w2[t];
  __syncthreads();
  for (int off = 64; off > 0; off >>= 1) { if (t < off) red[t] += red[t + off]; __syncthreads(); }
  if (t == 0) cand_val[e] = red[0] + b2[0] + gumbelf(0x9E3779B9u ^ (unsigned)idx);
}

__global__ void k_redmax(const float* __restrict__ vals, Ctrl* c1) {
  int n = (int)c1->n;
  for (int i = blockIdx.x * blockDim.x + threadIdx.x; i < n; i += gridDim.x * blockDim.x)
    atomicMaxFloat(&c1->maxv, vals[i]);
}
__global__ void k_redsum(const float* __restrict__ vals, Ctrl* c1) {
  int n = (int)c1->n;
  float m = c1->maxv;
  for (int i = blockIdx.x * blockDim.x + threadIdx.x; i < n; i += gridDim.x * blockDim.x)
    atomicAdd(&c1->sumexp, __expf(vals[i] - m));
}

__global__ void k_apply_drop(const float* __restrict__ g, const int* __restrict__ cand,
                             const float* __restrict__ cand_val, const int* __restrict__ sel,
                             float* __restrict__ adj, int* __restrict__ present,
                             int* __restrict__ last, const Ctrl* c0, const Ctrl* c1) {
  int n = (int)c0->target;
  float mx = c1->maxv, se = c1->sumexp;
  for (int e = blockIdx.x * blockDim.x + threadIdx.x; e < n; e += gridDim.x * blockDim.x) {
    int idx = cand[e];
    float soft = __expf(cand_val[e] - mx) / se;
    float hard = sel[e] ? 1.f : 0.f;
    float v = (hard - soft) + soft;                  // straight-through forward value
    adj[idx] = g[idx] * (1.f - v);
    if (sel[e]) {
      int s = idx >> 12;
      present[s] = 1;
      atomicMax(&last[s], idx);
    }
  }
}

__global__ void k_copy4(const float4* __restrict__ src, float4* __restrict__ dst, int n4) {
  for (int i = blockIdx.x * blockDim.x + threadIdx.x; i < n4; i += gridDim.x * blockDim.x)
    dst[i] = src[i];
}

__global__ void k_row_top8(const float* __restrict__ grad, const int* __restrict__ present,
                           int* __restrict__ top8) {
  int s = blockIdx.x;
  if (!present[s]) return;
  int t = threadIdx.x;                      // 256
  float lv[8]; int li[8];
#pragma unroll
  for (int k = 0; k < 8; ++k) { lv[k] = -3.0e38f; li[k] = 0; }
  for (int col = t; col < NN; col += 256) {
    float v = grad[(size_t)s * NN + col];
    if (v > lv[7]) {
      int k = 7;
      while (k > 0 && v > lv[k - 1]) { lv[k] = lv[k - 1]; li[k] = li[k - 1]; --k; }
      lv[k] = v; li[k] = col;
    }
  }
  __shared__ float sv[256 * 8];
  __shared__ int   si[256 * 8];
#pragma unroll
  for (int k = 0; k < 8; ++k) { sv[t * 8 + k] = lv[k]; si[t * 8 + k] = li[k]; }
  __syncthreads();
  if (t == 0) {
    float bv[8]; int bi[8];
#pragma unroll
    for (int k = 0; k < 8; ++k) { bv[k] = -3.0e38f; bi[k] = 0; }
    for (int j = 0; j < 256 * 8; ++j) {
      float v = sv[j];
      if (v > bv[7]) {
        int k = 7;
        while (k > 0 && v > bv[k - 1]) { bv[k] = bv[k - 1]; bi[k] = bi[k - 1]; --k; }
        bv[k] = v; bi[k] = si[j];
      }
    }
#pragma unroll
    for (int k = 0; k < 8; ++k) top8[s * 8 + k] = bi[k];
  }
}

__global__ void k_add_mlp(const float* __restrict__ src_emb, const float* __restrict__ dst_emb,
                          const float* __restrict__ w1, const float* __restrict__ b1,
                          const float* __restrict__ w2, const float* __restrict__ b2,
                          const int* __restrict__ cand, const int* __restrict__ sel,
                          const int* __restrict__ top8, const int* __restrict__ last,
                          float* __restrict__ adj, const Ctrl* c0) {
  int e = blockIdx.x;
  if (e >= (int)c0->target || !sel[e]) return;
  int idx = cand[e], s = idx >> 12;
  __shared__ float emb[2 * HH];
  __shared__ float red[HH];
  __shared__ float lg[8];
  int t = threadIdx.x;                      // 128
  emb[t] = src_emb[(size_t)s * HH + t];
  for (int k = 0; k < 8; ++k) {
    int d = top8[s * 8 + k];
    emb[HH + t] = dst_emb[(size_t)d * HH + t];
    __syncthreads();
    float h = b1[t];
    for (int j = 0; j < 2 * HH; ++j) h = fmaf(emb[j], w1[(size_t)j * HH + t], h);
    h = fmaxf(h, 0.f);
    red[t] = h * w2[t];
    __syncthreads();
    for (int off = 64; off > 0; off >>= 1) { if (t < off) red[t] += red[t + off]; __syncthreads(); }
    if (t == 0) lg[k] = red[0] + b2[0] + gumbelf(0xB5297A4Du ^ (unsigned)(idx * 8 + k));
    __syncthreads();
  }
  if (t == 0 && last[s] == idx) {           // sequential scatter: last write wins
    float m = -3.0e38f; int am = 0;
    for (int k = 0; k < 8; ++k) if (lg[k] > m) { m = lg[k]; am = k; }
    float ssum = 0.f; float p[8];
    for (int k = 0; k < 8; ++k) { p[k] = __expf(lg[k] - m); ssum += p[k]; }
    for (int k = 0; k < 8; ++k) {
      float soft = p[k] / ssum;
      float hard = (k == am) ? 1.f : 0.f;
      atomicAdd(&adj[(size_t)s * NN + top8[s * 8 + k]], (hard - soft) + soft);
    }
  }
}

__global__ void k_cvt16(const float* __restrict__ a, _Float16* __restrict__ o, int n) {
  for (int i = blockIdx.x * blockDim.x + threadIdx.x; i < n; i += gridDim.x * blockDim.x)
    o[i] = (_Float16)a[i];
}

// ---------------- metapath = adj @ adj.T ----------------
// Block tile 64 (M) x 128 (N); 8 waves in a 2x4 grid of 32x32 register tiles.
// K chunks of 64: TDM loads A rows (64x64 f16, 8KB) and B rows (128x64, 16KB)
// into LDS; waves consume via ds_load fragments; f16 WMMA, f32 accumulate.
#define KCHUNK 64
__global__ void k_metapath(const _Float16* __restrict__ A, float* __restrict__ C) {
  __shared__ _Float16 As[64 * KCHUNK];
  __shared__ _Float16 Bs[128 * KCHUNK];
  int lane = threadIdx.x & 31, wave = threadIdx.x >> 5;
  int mb = (int)blockIdx.x >> 5;            // 64 M blocks
  int nb = (int)blockIdx.x & 31;            // 32 N blocks
  int m_blk = mb * 64, n_blk = nb * 128;
  int wm = wave >> 2, wn = wave & 3;        // 2 x 4 waves
  int mo = wm * 32, no = wn * 32;           // offsets inside block tile
  v8f c00 = {}, c01 = {}, c10 = {}, c11 = {};
#if USE_TDM
  unsigned a_lds = (unsigned)(size_t)(void*)As;
  unsigned b_lds = (unsigned)(size_t)(void*)Bs;
#endif
  for (int k0 = 0; k0 < NN; k0 += KCHUNK) {
    __syncthreads();                         // LDS reuse fence
#if USE_TDM
    if (wave == 0) {
      tdm_load_2d(a_lds, A + (size_t)m_blk * NN + k0, KCHUNK, 64, NN);
      tdm_load_2d(b_lds, A + (size_t)n_blk * NN + k0, KCHUNK, 128, NN);
      __builtin_amdgcn_s_wait_tensorcnt(0);
    }
#else
    {  // cooperative fallback: 256 threads copy 64+128 rows of 64 f16
      int t = (int)threadIdx.x;
      int r = t >> 2, seg = (t & 3) * 16;    // 16 f16 = 32B per segment
      const _Float16* pa = A + (size_t)(m_blk + r) * NN + k0 + seg;
#pragma unroll
      for (int i = 0; i < 16; ++i) As[r * KCHUNK + seg + i] = pa[i];
#pragma unroll
      for (int j = 0; j < 2; ++j) {
        int rr = r + j * 64;
        const _Float16* pb = A + (size_t)(n_blk + rr) * NN + k0 + seg;
#pragma unroll
        for (int i = 0; i < 16; ++i) Bs[rr * KCHUNK + seg + i] = pb[i];
      }
    }
#endif
    __syncthreads();
#pragma unroll
    for (int kk = 0; kk < KCHUNK; kk += 32) {
      v16h a0 = load_A_h(As, KCHUNK, mo, kk, lane);
      v16h a1 = load_A_h(As, KCHUNK, mo + 16, kk, lane);
      v16h b0 = load_B_rows(Bs, KCHUNK, kk, no, lane);
      v16h b1 = load_B_rows(Bs, KCHUNK, kk, no + 16, lane);
      c00 = wmma16(a0, b0, c00);
      c01 = wmma16(a0, b1, c01);
      c10 = wmma16(a1, b0, c10);
      c11 = wmma16(a1, b1, c11);
    }
  }
  int m0 = m_blk + mo, n0 = n_blk + no;
  int col = lane & 15;
  int rb = (lane >> 4) * 8;
#pragma unroll
  for (int v = 0; v < 8; ++v) {
    C[(size_t)(m0 + rb + v) * NN + n0 + col]           = c00[v];
    C[(size_t)(m0 + rb + v) * NN + n0 + 16 + col]      = c01[v];
    C[(size_t)(m0 + 16 + rb + v) * NN + n0 + col]      = c10[v];
    C[(size_t)(m0 + 16 + rb + v) * NN + n0 + 16 + col] = c11[v];
  }
}

// ==================== launcher ====================
extern "C" void kernel_launch(void* const* d_in, const int* in_sizes, int n_in,
                              void* d_out, int out_size, void* d_ws, size_t ws_size,
                              hipStream_t stream) {
  if (ws_size < (size_t)WS_NEEDED) return;

  const float* src_emb = (const float*)d_in[0];
  const float* sub_g   = (const float*)d_in[1];
  const float* sub_gr  = (const float*)d_in[2];
  const float* ft_w    = (const float*)d_in[3];
  const float* ft_b    = (const float*)d_in[4];
  const float* dw1     = (const float*)d_in[5];
  const float* db1     = (const float*)d_in[6];
  const float* dw2     = (const float*)d_in[7];
  const float* db2     = (const float*)d_in[8];
  const float* aw1     = (const float*)d_in[9];
  const float* ab1     = (const float*)d_in[10];
  const float* aw2     = (const float*)d_in[11];
  const float* ab2     = (const float*)d_in[12];
  const int*   pK      = (const int*)d_in[13];

  char* ws = (char*)d_ws;
  _Float16* big16   = (_Float16*)(ws + OFF_BIG16);     // gT16, later adj16
  _Float16* featT   = (_Float16*)(ws + OFF_FEATT);
  float*    dst_emb = (float*)(ws + OFF_DSTEMB);
  int*      candidx = (int*)(ws + OFF_CANDIDX);
  float*    candval = (float*)(ws + OFF_CANDVAL);
  int*      sel     = (int*)(ws + OFF_SEL);
  int*      top8    = (int*)(ws + OFF_TOP8);
  int*      present = (int*)(ws + OFF_PRESENT);
  int*      last    = (int*)(ws + OFF_LAST);
  Ctrl*     c0      = (Ctrl*)(ws + OFF_CTRL);
  Ctrl*     c1      = c0 + 1;

  float* adj  = (float*)d_out;                  // output[0]: adj_adv (4096x4096)
  float* meta = (float*)d_out + (size_t)NTOT;   // output[1]: metapath

  // phase 0: init
  k_init_ctrl<<<1, 256, 0, stream>>>(c0, c1, pK);
  k_init_arrays<<<1024, 256, 0, stream>>>(sel, present, last);

  // phase 1: feature transform + dst embeddings (WMMA)
  k_feat<<<256, 256, 0, stream>>>(src_emb, ft_w, ft_b, featT);
  k_transpose<<<dim3(128, 128), dim3(32, 8), 0, stream>>>(sub_g, big16);
  k_dstemb<<<256, 256, 0, stream>>>(big16, featT, dst_emb);

  // phase 2: radix-select E2 most-negative existing-edge gradients
  const unsigned masks[4]  = {0x00000000u, 0xFF000000u, 0xFFFF0000u, 0xFFFFFF00u};
  const int      shifts[4] = {24, 16, 8, 0};
  for (int p = 0; p < 4; ++p) {
    k_hist<<<8192, 256, 0, stream>>>(sub_gr, sub_g, candval, c0, 0, masks[p], shifts[p], NTOT);
    k_scan<<<1, 256, 0, stream>>>(c0, shifts[p]);
  }
  k_gather<<<8192, 256, 0, stream>>>(sub_gr, sub_g, candval, c0, 0, NTOT, candidx, sel);

  // phase 3: drop MLP + gumbel, top-K of perturbed logits, straight-through drop
  k_copy4<<<4096, 256, 0, stream>>>((const float4*)sub_g, (float4*)adj, NTOT / 4);
  k_drop_mlp<<<MAXC2, 128, 0, stream>>>(src_emb, dst_emb, dw1, db1, dw2, db2, candidx, candval, c0);
  for (int p = 0; p < 4; ++p) {
    k_hist<<<512, 256, 0, stream>>>(sub_gr, sub_g, candval, c1, 1, masks[p], shifts[p], 0);
    k_scan<<<1, 256, 0, stream>>>(c1, shifts[p]);
  }
  k_gather<<<512, 256, 0, stream>>>(sub_gr, sub_g, candval, c1, 1, 0, candidx, sel);
  k_redmax<<<512, 256, 0, stream>>>(candval, c1);
  k_redsum<<<512, 256, 0, stream>>>(candval, c1);
  k_apply_drop<<<512, 256, 0, stream>>>(sub_g, candidx, candval, sel, adj, present, last, c0, c1);

  // phase 4: add branch
  k_row_top8<<<NN, 256, 0, stream>>>(sub_gr, present, top8);
  k_add_mlp<<<MAXC2, 128, 0, stream>>>(src_emb, dst_emb, aw1, ab1, aw2, ab2,
                                       candidx, sel, top8, last, adj, c0);

  // phase 5: metapath = adj @ adj.T  (TDM -> LDS -> f16 WMMA, f32 accumulate)
  k_cvt16<<<8192, 256, 0, stream>>>(adj, big16, NTOT);
  k_metapath<<<2048, 256, 0, stream>>>(big16, meta);
}
